// ProteinGraphEncoder_17583596110569
// MI455X (gfx1250) — matrix operand order repaired and checked by reading, not compiled, not run on hardware
//
#include <hip/hip_runtime.h>
#include <hip/hip_bf16.h>
#include <math.h>

// ---------------------------------------------------------------------------
// ProteinGraphEncoder forward for gfx1250 (MI455X).
// Dense GEMMs -> v_wmma_f32_16x16x32_f16 (f32 accum). Attention: streaming
// online-softmax per (head,query). Graph ops: float atomics.
// ---------------------------------------------------------------------------

#define NN 4096        // nodes
#define NE 98304       // edges
#define NF (NE + NN)   // edges + self loops = 102400
#define NG 16          // graphs
#define CHUNK 25600    // NF / 4

typedef __attribute__((ext_vector_type(16))) _Float16 v16h;
typedef __attribute__((ext_vector_type(8)))  float    v8f;

__device__ __forceinline__ float atomicMaxF(float* addr, float val) {
    int* ai = (int*)addr;
    int old = *ai;
    while (__int_as_float(old) < val) {
        int assumed = old;
        old = atomicCAS(ai, assumed, __float_as_int(val));
        if (old == assumed) break;
    }
    return __int_as_float(old);
}

__device__ __forceinline__ float eluf(float v) { return v > 0.f ? v : expm1f(v); }

__device__ __forceinline__ float actf(float v, int ACT) {
    if (ACT == 1) return fmaxf(v, 0.f);
    if (ACT == 2) return 0.5f * v * (1.f + erff(v * 0.70710678118654752f));
    return v;
}

// ---------------------------------------------------------------------------
// WMMA GEMM (K % 32 == 0 path): C[M,N] = act(A[M,K] @ B[K,N] + bias).
// f32 in memory, f16 compute, f32 accumulate. grid = (N/64, M/64), block=128
// (4 waves); each wave computes a 16x64 strip (4 WMMA tiles share one A frag).
// A-fragment (contiguous k order, not the ISA table order): lane l(0..15) row m0+l; hi=lane>>4;
//   elems j=0..15 -> k = kb + 16*hi + j
// B-fragment (same contiguous k order as A): lane l(0..15) col n; elems j -> k = kb+16*hi+j
// ---------------------------------------------------------------------------
template <int ACT>  // 0 none, 1 relu, 2 exact gelu
__global__ __launch_bounds__(128) void gemm_wmma_k(
    const float* __restrict__ A, const float* __restrict__ B,
    const float* __restrict__ bias, float* __restrict__ C,
    int M, int N, int K)
{
    __builtin_amdgcn_s_wait_tensorcnt(0);  // CDNA5 split-counter wait (no-op here)
    const int lane = threadIdx.x & 31;
    const int wave = threadIdx.x >> 5;
    const int m0 = blockIdx.y * 64 + wave * 16;
    const int n0 = blockIdx.x * 64;
    const int hi = lane >> 4;
    const int l16 = lane & 15;
    const float* Arow = A + (size_t)(m0 + l16) * K;

    v8f acc[4] = {};

    for (int kb = 0; kb < K; kb += 32) {
        __builtin_prefetch(Arow + kb + 64, 0, 1);
        v16h af;
        const float4 x0 = *(const float4*)(Arow + kb + 16 * hi);
        const float4 x1 = *(const float4*)(Arow + kb + 16 * hi + 4);
        const float4 x2 = *(const float4*)(Arow + kb + 16 * hi + 8);
        const float4 x3 = *(const float4*)(Arow + kb + 16 * hi + 12);
        af[0]  = (_Float16)x0.x; af[1]  = (_Float16)x0.y;
        af[2]  = (_Float16)x0.z; af[3]  = (_Float16)x0.w;
        af[4]  = (_Float16)x1.x; af[5]  = (_Float16)x1.y;
        af[6]  = (_Float16)x1.z; af[7]  = (_Float16)x1.w;
        af[8]  = (_Float16)x2.x; af[9]  = (_Float16)x2.y;
        af[10] = (_Float16)x2.z; af[11] = (_Float16)x2.w;
        af[12] = (_Float16)x3.x; af[13] = (_Float16)x3.y;
        af[14] = (_Float16)x3.z; af[15] = (_Float16)x3.w;
#pragma unroll
        for (int nt = 0; nt < 4; nt++) {
            const int col = n0 + nt * 16 + l16;
            const float* Bp = B + (size_t)(kb + 16 * hi) * N + col;
            v16h bf;
#pragma unroll
            for (int j = 0; j < 16; j++) bf[j] = (_Float16)Bp[(size_t)j * N];
            acc[nt] = __builtin_amdgcn_wmma_f32_16x16x32_f16(
                false, af, false, bf, (short)0, acc[nt], false, false);
        }
    }

#pragma unroll
    for (int nt = 0; nt < 4; nt++) {
        const int col = n0 + nt * 16 + l16;
        const float bv = bias ? bias[col] : 0.f;
#pragma unroll
        for (int r = 0; r < 8; r++)
            C[(size_t)(m0 + r + 8 * hi) * N + col] = actf(acc[nt][r] + bv, ACT);
    }
}

// Tail variant for K < 32 (oh_fc K=21, aa_fc K=9): single zero-padded k-step.
template <int ACT>
__global__ __launch_bounds__(128) void gemm_wmma_tail_k(
    const float* __restrict__ A, const float* __restrict__ B,
    const float* __restrict__ bias, float* __restrict__ C,
    int M, int N, int K)
{
    const int lane = threadIdx.x & 31;
    const int wave = threadIdx.x >> 5;
    const int m0 = blockIdx.y * 64 + wave * 16;
    const int n0 = blockIdx.x * 64;
    const int hi = lane >> 4;
    const int l16 = lane & 15;
    const float* Arow = A + (size_t)(m0 + l16) * K;

    v16h af;
#pragma unroll
    for (int j = 0; j < 8; j++) {
        const int k1 = 16 * hi + j, k2 = 16 * hi + 8 + j;
        af[j]     = (_Float16)((k1 < K) ? Arow[k1] : 0.f);
        af[j + 8] = (_Float16)((k2 < K) ? Arow[k2] : 0.f);
    }
    v8f acc[4] = {};
#pragma unroll
    for (int nt = 0; nt < 4; nt++) {
        const int col = n0 + nt * 16 + l16;
        v16h bf;
#pragma unroll
        for (int j = 0; j < 16; j++) {
            const int k = 16 * hi + j;
            bf[j] = (_Float16)((k < K) ? B[(size_t)k * N + col] : 0.f);
        }
        acc[nt] = __builtin_amdgcn_wmma_f32_16x16x32_f16(
            false, af, false, bf, (short)0, acc[nt], false, false);
    }
#pragma unroll
    for (int nt = 0; nt < 4; nt++) {
        const int col = n0 + nt * 16 + l16;
        const float bv = bias ? bias[col] : 0.f;
#pragma unroll
        for (int r = 0; r < 8; r++)
            C[(size_t)(m0 + r + 8 * hi) * N + col] = actf(acc[nt][r] + bv, ACT);
    }
}

// ---------------------------------------------------------------------------
// small utility kernels
// ---------------------------------------------------------------------------
__global__ void fill_k(float* p, size_t n, float v) {
    size_t t = (size_t)blockIdx.x * blockDim.x + threadIdx.x;
    if (t < n) p[t] = v;
}

__global__ void gather_pos_k(float* feat, const float* pe, const int* pos, size_t total) {
    size_t t = (size_t)blockIdx.x * blockDim.x + threadIdx.x;
    if (t >= total) return;
    int n = (int)(t >> 5), j = (int)(t & 31);
    feat[(size_t)n * 416 + j] = pe[(size_t)pos[n] * 32 + j];
}

__global__ void copy_cols_k(float* feat, const float* src, int rows, int cols, int off) {
    size_t t = (size_t)blockIdx.x * blockDim.x + threadIdx.x;
    if (t >= (size_t)rows * cols) return;
    int r = (int)(t / cols), j = (int)(t % cols);
    feat[(size_t)r * 416 + off + j] = src[t];
}

// BatchNorm over rows (training stats), per column; two-pass, optional relu.
__global__ __launch_bounds__(256) void bn_cols_k(
    const float* __restrict__ x, float* __restrict__ outp,
    const float* g, const float* b, int rows, int C, int relu)
{
    const int c = blockIdx.x, tid = threadIdx.x;
    __shared__ float sd[256];
    __shared__ float stat;
    float s = 0.f;
    for (int r = tid; r < rows; r += 256) s += x[(size_t)r * C + c];
    sd[tid] = s; __syncthreads();
    for (int o = 128; o > 0; o >>= 1) { if (tid < o) sd[tid] += sd[tid + o]; __syncthreads(); }
    if (tid == 0) stat = sd[0] / rows;
    __syncthreads();
    const float mean = stat;
    float s2 = 0.f;
    for (int r = tid; r < rows; r += 256) { float d = x[(size_t)r * C + c] - mean; s2 += d * d; }
    __syncthreads();
    sd[tid] = s2; __syncthreads();
    for (int o = 128; o > 0; o >>= 1) { if (tid < o) sd[tid] += sd[tid + o]; __syncthreads(); }
    if (tid == 0) stat = rsqrtf(sd[0] / rows + 1e-5f);
    __syncthreads();
    const float rstd = stat, gg = g[c], bb = b[c];
    for (int r = tid; r < rows; r += 256) {
        float v = (x[(size_t)r * C + c] - mean) * rstd * gg + bb;
        if (relu) v = fmaxf(v, 0.f);
        outp[(size_t)r * C + c] = v;
    }
}

// LayerNorm over 128 dims; one wave per row; optional fused residual add.
__global__ __launch_bounds__(128) void ln128_k(
    const float* __restrict__ x, const float* __restrict__ y,
    float* __restrict__ outp, const float* g, const float* b, int rows)
{
    const int lane = threadIdx.x & 31, wv = threadIdx.x >> 5;
    const int row = blockIdx.x * 4 + wv;
    if (row >= rows) return;
    const float* xr = x + (size_t)row * 128;
    const float* yr = y ? y + (size_t)row * 128 : nullptr;
    float v[4]; float s = 0.f;
#pragma unroll
    for (int i = 0; i < 4; i++) {
        int c = lane + 32 * i;
        float t = xr[c] + (yr ? yr[c] : 0.f);
        v[i] = t; s += t;
    }
#pragma unroll
    for (int o = 16; o > 0; o >>= 1) s += __shfl_xor(s, o, 32);
    const float mean = s * (1.f / 128.f);
    float s2 = 0.f;
#pragma unroll
    for (int i = 0; i < 4; i++) { float d = v[i] - mean; s2 += d * d; }
#pragma unroll
    for (int o = 16; o > 0; o >>= 1) s2 += __shfl_xor(s2, o, 32);
    const float rstd = rsqrtf(s2 * (1.f / 128.f) + 1e-5f);
    float* orow = outp + (size_t)row * 128;
#pragma unroll
    for (int i = 0; i < 4; i++) {
        int c = lane + 32 * i;
        orow[c] = (v[i] - mean) * rstd * g[c] + b[c];
    }
}

// Streaming attention: one thread per (head, query). qkv: [N,384] (q|k|v).
__global__ void attn_k(const float* __restrict__ qkv, float* __restrict__ outp, int n, size_t total) {
    size_t t = (size_t)blockIdx.x * blockDim.x + threadIdx.x;
    if (t >= total) return;
    const int h = (int)(t / n), q_n = (int)(t % n);
    const int ko = 128 + h * 32, vo = 256 + h * 32;
    float4 q[8];
    const float4* qp = (const float4*)(qkv + (size_t)q_n * 384 + h * 32);
#pragma unroll
    for (int i = 0; i < 8; i++) q[i] = qp[i];
    float m = -3.4e38f, l = 0.f;
    float4 acc[8];
#pragma unroll
    for (int i = 0; i < 8; i++) acc[i] = make_float4(0.f, 0.f, 0.f, 0.f);
    for (int j = 0; j < n; j++) {
        const float4* kp = (const float4*)(qkv + (size_t)j * 384 + ko);
        float s = 0.f;
#pragma unroll
        for (int i = 0; i < 8; i++) {
            float4 kv = kp[i];
            s += q[i].x * kv.x + q[i].y * kv.y + q[i].z * kv.z + q[i].w * kv.w;
        }
        s *= 0.1767766952966369f;  // 1/sqrt(32)
        float p;
        if (s > m) {
            const float sc = __expf(m - s);
            l *= sc;
#pragma unroll
            for (int i = 0; i < 8; i++) { acc[i].x *= sc; acc[i].y *= sc; acc[i].z *= sc; acc[i].w *= sc; }
            m = s; p = 1.f;
        } else {
            p = __expf(s - m);
        }
        l += p;
        const float4* vp = (const float4*)(qkv + (size_t)j * 384 + vo);
#pragma unroll
        for (int i = 0; i < 8; i++) {
            float4 vv = vp[i];
            acc[i].x += p * vv.x; acc[i].y += p * vv.y;
            acc[i].z += p * vv.z; acc[i].w += p * vv.w;
        }
    }
    const float rl = 1.f / l;
    float4* op = (float4*)(outp + (size_t)q_n * 128 + h * 32);
#pragma unroll
    for (int i = 0; i < 8; i++)
        op[i] = make_float4(acc[i].x * rl, acc[i].y * rl, acc[i].z * rl, acc[i].w * rl);
}

// edge MLP layer 1: ea1[e,j] = relu(w[e]*W[j] + b[j]), j<64
__global__ void edge1_k(float* ea1, const float* ew, const float* W, const float* b, size_t total) {
    size_t t = (size_t)blockIdx.x * blockDim.x + threadIdx.x;
    if (t >= total) return;
    const int e = (int)(t >> 6), j = (int)(t & 63);
    const float v = ew[e] * W[j] + b[j];
    ea1[t] = v > 0.f ? v : 0.f;
}

__global__ void edge_cnt_k(float* cnt, const int* dst0, size_t total) {
    size_t t = (size_t)blockIdx.x * blockDim.x + threadIdx.x;
    if (t < total) atomicAdd(&cnt[dst0[t]], 1.f);
}

__global__ void loop_sum_k(float* ea_full, const int* dst0, size_t total) {
    size_t t = (size_t)blockIdx.x * blockDim.x + threadIdx.x;
    if (t >= total) return;
    const int e = (int)(t >> 7), c = (int)(t & 127);
    atomicAdd(&ea_full[((size_t)NE + dst0[e]) * 128 + c], ea_full[(size_t)e * 128 + c]);
}

__global__ void loop_div_k(float* ea_full, const float* cnt, size_t total) {
    size_t t = (size_t)blockIdx.x * blockDim.x + threadIdx.x;
    if (t >= total) return;
    const int n = (int)(t >> 7), c = (int)(t & 127);
    ea_full[((size_t)NE + n) * 128 + c] *= 1.f / fmaxf(cnt[n], 1.f);
}

__global__ void initbias_k(float* outp, const float* bias, size_t total) {
    size_t t = (size_t)blockIdx.x * blockDim.x + threadIdx.x;
    if (t < total) outp[t] = bias[t & 511];
}

// GATv2 per-edge logits: block = 128 threads, one edge per block.
__global__ __launch_bounds__(128) void gat_logit_k(
    const float* __restrict__ eaW, const float* __restrict__ xl,
    const float* __restrict__ xr, const float* __restrict__ att,
    const int* __restrict__ src0, const int* __restrict__ dst0,
    float* __restrict__ logits, float* __restrict__ segm, int chunkStart)
{
    const int i = blockIdx.x;
    const int ge = chunkStart + i;
    const int c = threadIdx.x;
    int s, d;
    if (ge < NE) { s = src0[ge]; d = dst0[ge]; } else { s = ge - NE; d = s; }
    __shared__ float red[4][128];
#pragma unroll
    for (int h = 0; h < 4; h++) {
        float t = xl[(size_t)s * 512 + h * 128 + c] + xr[(size_t)d * 512 + h * 128 + c]
                + eaW[(size_t)i * 512 + h * 128 + c];
        t = t > 0.f ? t : 0.2f * t;  // leaky_relu(0.2)
        red[h][c] = t * att[h * 128 + c];
    }
    __syncthreads();
    for (int o = 64; o > 0; o >>= 1) {
        if (c < o) {
#pragma unroll
            for (int h = 0; h < 4; h++) red[h][c] += red[h][c + o];
        }
        __syncthreads();
    }
    if (c < 4) {
        const float lg = red[c][0];
        logits[(size_t)ge * 4 + c] = lg;
        atomicMaxF(&segm[(size_t)d * 4 + c], lg);
    }
}

__global__ void gat_expsum_k(float* logits, const float* segm, float* segs,
                             const int* dst0, size_t total) {
    size_t t = (size_t)blockIdx.x * blockDim.x + threadIdx.x;
    if (t >= total) return;
    const int ge = (int)(t >> 2), h = (int)(t & 3);
    const int d = (ge < NE) ? dst0[ge] : ge - NE;
    float m = segm[(size_t)d * 4 + h];
    if (!(m > -1e37f)) m = 0.f;  // isfinite clamp (init = -3.4e38)
    const float a = __expf(logits[t] - m);
    logits[t] = a;
    atomicAdd(&segs[(size_t)d * 4 + h], a);
}

__global__ void gat_agg_k(const float* __restrict__ logits, const float* __restrict__ segs,
                          const float* __restrict__ xl, const int* __restrict__ src0,
                          const int* __restrict__ dst0, float* __restrict__ outp, size_t total) {
    size_t t = (size_t)blockIdx.x * blockDim.x + threadIdx.x;
    if (t >= total) return;
    const int c = (int)(t & 511);
    const int ge = (int)(t >> 9);
    int s, d;
    if (ge < NE) { s = src0[ge]; d = dst0[ge]; } else { s = ge - NE; d = s; }
    const int h = c >> 7;
    const float a = logits[(size_t)ge * 4 + h];
    const float sum = segs[(size_t)d * 4 + h];
    const float alpha = a / fmaxf(sum, 1e-16f);
    atomicAdd(&outp[(size_t)d * 512 + c], xl[(size_t)s * 512 + c] * alpha);
}

// GraphNorm: mode 1 = elu(out), mode 2 = elu(out + resid)
__global__ __launch_bounds__(256) void graphnorm_k(
    const float* __restrict__ x, const float* __restrict__ resid,
    float* __restrict__ outp, const float* g, const float* b, const float* ms,
    int rows, int C, int mode)
{
    const int c = blockIdx.x, tid = threadIdx.x;
    __shared__ float sd[256];
    __shared__ float stat;
    float s = 0.f;
    for (int r = tid; r < rows; r += 256) s += x[(size_t)r * C + c];
    sd[tid] = s; __syncthreads();
    for (int o = 128; o > 0; o >>= 1) { if (tid < o) sd[tid] += sd[tid + o]; __syncthreads(); }
    if (tid == 0) stat = sd[0] / rows;
    __syncthreads();
    const float mm = stat * ms[c];
    float s2 = 0.f;
    for (int r = tid; r < rows; r += 256) { float d = x[(size_t)r * C + c] - mm; s2 += d * d; }
    __syncthreads();
    sd[tid] = s2; __syncthreads();
    for (int o = 128; o > 0; o >>= 1) { if (tid < o) sd[tid] += sd[tid + o]; __syncthreads(); }
    if (tid == 0) stat = rsqrtf(sd[0] / rows + 1e-5f);
    __syncthreads();
    const float rstd = stat, gg = g[c], bb = b[c];
    for (int r = tid; r < rows; r += 256) {
        float v = (x[(size_t)r * C + c] - mm) * rstd * gg + bb;
        if (mode == 1) v = eluf(v);
        else if (mode == 2) v = eluf(v + resid[(size_t)r * C + c]);
        outp[(size_t)r * C + c] = v;
    }
}

__global__ void scores_k(const float* __restrict__ h, const float* w2, const float* b2,
                         float* sc, size_t total) {
    size_t t = (size_t)blockIdx.x * blockDim.x + threadIdx.x;
    if (t >= total) return;
    float acc = b2[0];
    for (int k = 0; k < 128; k++) acc += h[t * 128 + k] * w2[k];
    sc[t] = acc;
}

__global__ __launch_bounds__(1024) void softmax_all_k(const float* s, float* w, int n) {
    __shared__ float sd[1024];
    const int tid = threadIdx.x;
    float m = -3.4e38f;
    for (int i = tid; i < n; i += 1024) m = fmaxf(m, s[i]);
    sd[tid] = m; __syncthreads();
    for (int o = 512; o > 0; o >>= 1) { if (tid < o) sd[tid] = fmaxf(sd[tid], sd[tid + o]); __syncthreads(); }
    m = sd[0]; __syncthreads();
    float sum = 0.f;
    for (int i = tid; i < n; i += 1024) sum += __expf(s[i] - m);
    sd[tid] = sum; __syncthreads();
    for (int o = 512; o > 0; o >>= 1) { if (tid < o) sd[tid] += sd[tid + o]; __syncthreads(); }
    const float inv = 1.f / sd[0];
    for (int i = tid; i < n; i += 1024) w[i] = __expf(s[i] - m) * inv;
}

__global__ void cnts_k(float* cnts, const int* batch, size_t total) {
    size_t t = (size_t)blockIdx.x * blockDim.x + threadIdx.x;
    if (t < total) atomicAdd(&cnts[batch[t]], 1.f);
}

__global__ void poolacc_k(const float* __restrict__ x, const float* __restrict__ wv,
                          const int* __restrict__ batch, float* ap, float* sp, float* mp,
                          size_t total) {
    size_t t = (size_t)blockIdx.x * blockDim.x + threadIdx.x;
    if (t >= total) return;
    const int n = (int)(t >> 9), c = (int)(t & 511);
    const int g = batch[n];
    const float xv = x[t];
    atomicAdd(&ap[(size_t)g * 512 + c], xv * wv[n]);
    atomicAdd(&sp[(size_t)g * 512 + c], xv);
    atomicMaxF(&mp[(size_t)g * 512 + c], xv);
}

__global__ void assemble_k(float* pin, const float* ap, const float* mp,
                           const float* sp, const float* cn, size_t total) {
    size_t t = (size_t)blockIdx.x * blockDim.x + threadIdx.x;
    if (t >= total) return;
    const int g = (int)(t / 1536), k = (int)(t % 1536);
    float v;
    if (k < 512)       v = ap[(size_t)g * 512 + k];
    else if (k < 1024) v = mp[(size_t)g * 512 + (k - 512)];
    else               v = sp[(size_t)g * 512 + (k - 1024)] / fmaxf(cn[g], 1.f);
    pin[t] = v;
}

__global__ __launch_bounds__(128) void pool_proj_k(const float* __restrict__ pin,
                                                   const float* __restrict__ Wp,
                                                   const float* __restrict__ bp,
                                                   float* __restrict__ outp) {
    const int g = blockIdx.x, j = threadIdx.x;
    float acc = bp[j];
    for (int k = 0; k < 1536; k++) acc += pin[(size_t)g * 1536 + k] * Wp[(size_t)k * 128 + j];
    outp[(size_t)g * 128 + j] = acc;
}

// ---------------------------------------------------------------------------
// host launch
// ---------------------------------------------------------------------------
#define L1D(kern, total, blk, ...)                                              \
    do {                                                                        \
        long long _t = (long long)(total);                                      \
        dim3 _g((unsigned)((_t + (blk)-1) / (blk)));                            \
        kern<<<_g, dim3(blk), 0, stream>>>(__VA_ARGS__);                        \
    } while (0)

extern "C" void kernel_launch(void* const* d_in, const int* in_sizes, int n_in,
                              void* d_out, int out_size, void* d_ws, size_t ws_size,
                              hipStream_t stream)
{
    (void)in_sizes; (void)n_in; (void)out_size; (void)ws_size;
    // ---- input mapping (setup_inputs insertion order; params pytree in
    //      _make_params insertion order) ----
    const int*   pos    = (const int*)  d_in[0];
    const float* evo    = (const float*)d_in[1];
    const float* onehot = (const float*)d_in[2];
    const float* aa     = (const float*)d_in[3];
    const float* ew     = (const float*)d_in[4];
    const int*   eidx   = (const int*)  d_in[5];
    const int*   batch  = (const int*)  d_in[6];
    const int* src0 = eidx;
    const int* dst0 = eidx + NE;
    auto P = [&](int i) { return (const float*)d_in[i]; };
    // 7 pos_embed | 8/9 evo_fc1 W,b | 10/11 evo_fc2 | 12/13 evo_fc_bn g,b
    // tr layer L (base 14+12L): +0/1 Wqkv, +2/3 Wo, +4/5 ln1, +6/7 W1, +8/9 W2, +10/11 ln2
    // 38/39 evo_ln | 40/41 evo_bn | 42/43 oh_fc | 44/45 oh_bn | 46/47 aa_fc | 48/49 aa_bn
    // 50/51 edge1 | 52/53 edge2
    // gat1 base 54: Wl.W,Wl.b,Wr.W,Wr.b,We,att,bias | 61-63 gn1 g,b,ms
    // gat2 base 64 | 71-73 gn2 | 74/75 res_proj | 76/77 pool_a1 | 78/79 pool_a2
    // 80/81 pool_proj | 82/83 raw_proj

    // ---- workspace layout ----
    float* Wf = (float*)d_ws;
    size_t off = 0;
    auto alloc = [&](size_t n) { float* p = Wf + off; off += n; return p; };
    float* feat    = alloc((size_t)NN * 416);
    float* ev0     = alloc((size_t)NN * 256);
    float* tA      = alloc((size_t)NN * 128);
    float* tB      = alloc((size_t)NN * 128);
    float* tC      = alloc((size_t)NN * 128);
    float* qkv     = alloc((size_t)NN * 384);
    float* ff1     = alloc((size_t)NN * 512);
    float* ea_full = alloc((size_t)NF * 128);
    float* ea1     = alloc((size_t)NE * 64);
    float* cnt     = alloc(NN);
    float* eaW     = alloc((size_t)CHUNK * 512);
    float* xl      = alloc((size_t)NN * 512);
    float* xr      = alloc((size_t)NN * 512);
    float* logits  = alloc((size_t)NF * 4);
    float* segm    = alloc((size_t)NN * 4);
    float* segs    = alloc((size_t)NN * 4);
    float* gout    = alloc((size_t)NN * 512);
    float* g1      = alloc((size_t)NN * 512);
    float* res     = alloc((size_t)NN * 512);
    float* xfin    = alloc((size_t)NN * 512);
    float* scoresb = alloc(NN);
    float* wsm     = alloc(NN);
    float* attn_p  = alloc((size_t)NG * 512);
    float* max_p   = alloc((size_t)NG * 512);
    float* sum_p   = alloc((size_t)NG * 512);
    float* cnts    = alloc(NG);
    float* pin     = alloc((size_t)NG * 1536);

    auto gemm = [&](const float* A, const float* Bm, const float* bias, float* C,
                    int M, int N, int K, int act) {
        dim3 g(N / 64, M / 64), blk(128);
        if ((K & 31) == 0) {
            if (act == 1)      gemm_wmma_k<1><<<g, blk, 0, stream>>>(A, Bm, bias, C, M, N, K);
            else if (act == 2) gemm_wmma_k<2><<<g, blk, 0, stream>>>(A, Bm, bias, C, M, N, K);
            else               gemm_wmma_k<0><<<g, blk, 0, stream>>>(A, Bm, bias, C, M, N, K);
        } else {
            if (act == 1)      gemm_wmma_tail_k<1><<<g, blk, 0, stream>>>(A, Bm, bias, C, M, N, K);
            else if (act == 2) gemm_wmma_tail_k<2><<<g, blk, 0, stream>>>(A, Bm, bias, C, M, N, K);
            else               gemm_wmma_tail_k<0><<<g, blk, 0, stream>>>(A, Bm, bias, C, M, N, K);
        }
    };

    // ---- node features ----
    L1D(gather_pos_k, (size_t)NN * 32, 256, feat, P(7), pos, (size_t)NN * 32);
    gemm(evo, P(8), P(9), ev0, NN, 256, 1280, 1);              // evo_fc1 + relu
    gemm(ev0, P(10), P(11), tA, NN, 128, 256, 0);              // evo_fc2
    bn_cols_k<<<128, 256, 0, stream>>>(tA, tB, P(12), P(13), NN, 128, 1);

    for (int L = 0; L < 2; L++) {                              // transformer x2
        const int pb = 14 + 12 * L;
        gemm(tB, P(pb + 0), P(pb + 1), qkv, NN, 384, 128, 0);
        L1D(attn_k, (size_t)NN * 4, 256, qkv, tA, NN, (size_t)NN * 4);
        gemm(tA, P(pb + 2), P(pb + 3), tC, NN, 128, 128, 0);
        ln128_k<<<NN / 4, 128, 0, stream>>>(tB, tC, tB, P(pb + 4), P(pb + 5), NN);
        gemm(tB, P(pb + 6), P(pb + 7), ff1, NN, 512, 128, 2);  // W1 + exact gelu
        gemm(ff1, P(pb + 8), P(pb + 9), tC, NN, 128, 512, 0);
        ln128_k<<<NN / 4, 128, 0, stream>>>(tB, tC, tB, P(pb + 10), P(pb + 11), NN);
    }
    ln128_k<<<NN / 4, 128, 0, stream>>>(tB, nullptr, tA, P(38), P(39), NN);
    bn_cols_k<<<128, 256, 0, stream>>>(tA, tB, P(40), P(41), NN, 128, 0);
    L1D(copy_cols_k, (size_t)NN * 128, 256, feat, tB, NN, 128, 32);

    gemm(onehot, P(42), P(43), tA, NN, 128, 21, 0);
    bn_cols_k<<<128, 256, 0, stream>>>(tA, tB, P(44), P(45), NN, 128, 1);
    L1D(copy_cols_k, (size_t)NN * 128, 256, feat, tB, NN, 128, 160);

    gemm(aa, P(46), P(47), tA, NN, 128, 9, 0);
    bn_cols_k<<<128, 256, 0, stream>>>(tA, tB, P(48), P(49), NN, 128, 1);
    L1D(copy_cols_k, (size_t)NN * 128, 256, feat, tB, NN, 128, 288);

    // ---- edge features + self-loop mean fill ----
    L1D(edge1_k, (size_t)NE * 64, 256, ea1, ew, P(50), P(51), (size_t)NE * 64);
    gemm(ea1, P(52), P(53), ea_full, NE, 128, 64, 0);
    L1D(fill_k, (size_t)NN * 128, 256, ea_full + (size_t)NE * 128, (size_t)NN * 128, 0.f);
    L1D(fill_k, (size_t)NN, 256, cnt, (size_t)NN, 0.f);
    L1D(edge_cnt_k, (size_t)NE, 256, cnt, dst0, (size_t)NE);
    L1D(loop_sum_k, (size_t)NE * 128, 256, ea_full, dst0, (size_t)NE * 128);
    L1D(loop_div_k, (size_t)NN * 128, 256, ea_full, cnt, (size_t)NN * 128);

    // ---- GATv2 layer (chunked We GEMM to bound scratch) ----
    auto gat = [&](const float* xin, int D, int pb) {
        gemm(xin, P(pb + 0), P(pb + 1), xl, NN, 512, D, 0);
        gemm(xin, P(pb + 2), P(pb + 3), xr, NN, 512, D, 0);
        L1D(fill_k, (size_t)NN * 4, 256, segm, (size_t)NN * 4, -3.4e38f);
        L1D(fill_k, (size_t)NN * 4, 256, segs, (size_t)NN * 4, 0.f);
        L1D(initbias_k, (size_t)NN * 512, 256, gout, P(pb + 6), (size_t)NN * 512);
        for (int ch = 0; ch < 4; ch++) {
            const int cs = ch * CHUNK;
            gemm(ea_full + (size_t)cs * 128, P(pb + 4), nullptr, eaW, CHUNK, 512, 128, 0);
            gat_logit_k<<<CHUNK, 128, 0, stream>>>(eaW, xl, xr, P(pb + 5), src0, dst0,
                                                   logits, segm, cs);
        }
        L1D(gat_expsum_k, (size_t)NF * 4, 256, logits, segm, segs, dst0, (size_t)NF * 4);
        L1D(gat_agg_k, (size_t)NF * 512, 256, logits, segs, xl, src0, dst0, gout,
            (size_t)NF * 512);
    };

    gat(feat, 416, 54);
    graphnorm_k<<<512, 256, 0, stream>>>(gout, nullptr, g1, P(61), P(62), P(63), NN, 512, 1);
    gemm(feat, P(74), P(75), res, NN, 512, 416, 0);
    gat(g1, 512, 64);
    graphnorm_k<<<512, 256, 0, stream>>>(gout, res, xfin, P(71), P(72), P(73), NN, 512, 2);

    // ---- pooling + outputs ----
    gemm(xfin, P(76), P(77), tA, NN, 128, 512, 1);             // pool_a1 + relu
    L1D(scores_k, (size_t)NN, 256, tA, P(78), P(79), scoresb, (size_t)NN);
    softmax_all_k<<<1, 1024, 0, stream>>>(scoresb, wsm, NN);
    L1D(fill_k, (size_t)NG * 512, 256, attn_p, (size_t)NG * 512, 0.f);
    L1D(fill_k, (size_t)NG * 512, 256, sum_p, (size_t)NG * 512, 0.f);
    L1D(fill_k, (size_t)NG * 512, 256, max_p, (size_t)NG * 512, -3.4e38f);
    L1D(fill_k, (size_t)NG, 32, cnts, (size_t)NG, 0.f);
    L1D(cnts_k, (size_t)NN, 256, cnts, batch, (size_t)NN);
    L1D(poolacc_k, (size_t)NN * 512, 256, xfin, wsm, batch, attn_p, sum_p, max_p,
        (size_t)NN * 512);
    L1D(assemble_k, (size_t)NG * 1536, 256, pin, attn_p, max_p, sum_p, cnts,
        (size_t)NG * 1536);
    pool_proj_k<<<NG, 128, 0, stream>>>(pin, P(80), P(81), (float*)d_out);
    gemm(xfin, P(82), P(83), ((float*)d_out) + (size_t)NG * 128, NN, 128, 512, 0);
}
